// LSTMWebNet_31379031065222
// MI455X (gfx1250) — matrix operand order repaired and checked by reading, not compiled, not run on hardware
//
#include <hip/hip_runtime.h>

// ---------------------------------------------------------------------------
// Problem constants (match reference)
// ---------------------------------------------------------------------------
constexpr int kB  = 1024;   // batch
constexpr int kNI = 2048;
constexpr int kNH = 4096;
constexpr int kNO = 1024;
constexpr int kC  = 4;      // LSTM cell width

// ---------------------------------------------------------------------------
// WMMA types (CDNA5 / gfx1250, wave32)
// ---------------------------------------------------------------------------
typedef __attribute__((ext_vector_type(16))) __bf16 v16bf;
typedef __attribute__((ext_vector_type(8)))  float  v8f;
typedef __attribute__((ext_vector_type(4)))  int    v4i;

union FragAB {
    v16bf v;
    uint4 q[2];   // 2 x 16 bytes = 16 bf16
};

// ---------------------------------------------------------------------------
// fp32 -> bf16 (RNE)
// ---------------------------------------------------------------------------
__device__ __forceinline__ unsigned short f32_to_bf16(float f) {
    unsigned int u = __float_as_uint(f);
    unsigned int r = 0x7FFFu + ((u >> 16) & 1u);
    return (unsigned short)((u + r) >> 16);
}

__device__ __forceinline__ unsigned int pack2_bf16(float lo, float hi) {
#if __has_builtin(__builtin_amdgcn_cvt_pk_bf16_f32)
    typedef __attribute__((ext_vector_type(2))) __bf16 v2bf;
    v2bf p = __builtin_amdgcn_cvt_pk_bf16_f32(lo, hi);
    return __builtin_bit_cast(unsigned int, p);
#else
    return (unsigned)f32_to_bf16(lo) | ((unsigned)f32_to_bf16(hi) << 16);
#endif
}

__device__ __forceinline__ uint2 pack4_bf16(float4 v) {
    uint2 p;
    p.x = pack2_bf16(v.x, v.y);
    p.y = pack2_bf16(v.z, v.w);
    return p;
}

// ---------------------------------------------------------------------------
// Bulk fp32 -> bf16 conversion, 8 elems / thread (bandwidth-bound)
// ---------------------------------------------------------------------------
__global__ __launch_bounds__(256)
void convert_bf16(const float* __restrict__ in, unsigned short* __restrict__ outp, int n8)
{
    int idx = blockIdx.x * 256 + threadIdx.x;
    if (idx >= n8) return;
    const float4* p = (const float4*)in + (size_t)idx * 2;
    float4 a = p[0];
    float4 b = p[1];
    uint4 r;
    r.x = pack2_bf16(a.x, a.y);
    r.y = pack2_bf16(a.z, a.w);
    r.z = pack2_bf16(b.x, b.y);
    r.w = pack2_bf16(b.z, b.w);
    ((uint4*)outp)[idx] = r;
}

// ---------------------------------------------------------------------------
// Async global -> LDS 16-byte copy (CDNA5 ASYNCcnt path) with safe fallback.
// Builtin signature (probe-verified): (v4i32 AS1*, v4i32 AS3*, imm, imm)
// ---------------------------------------------------------------------------
#if __has_builtin(__builtin_amdgcn_global_load_async_to_lds_b128)
#define HAVE_ASYNC_LDS 1
typedef v4i __attribute__((address_space(1)))* g_v4i_ptr;
typedef v4i __attribute__((address_space(3)))* l_v4i_ptr;
#else
#define HAVE_ASYNC_LDS 0
#endif

__device__ __forceinline__ void copy16_g2l(const unsigned short* __restrict__ g,
                                           unsigned short* __restrict__ l)
{
#if HAVE_ASYNC_LDS
    __builtin_amdgcn_global_load_async_to_lds_b128(
        (g_v4i_ptr)g, (l_v4i_ptr)l, /*offset=*/0, /*cpol=*/0);
#else
    *(uint4*)l = *(const uint4*)g;
#endif
}

// wait until at most N of this wave's async copies remain in flight
__device__ __forceinline__ void wait_async_le6()
{
#if HAVE_ASYNC_LDS
 #if __has_builtin(__builtin_amdgcn_s_wait_asynccnt)
    __builtin_amdgcn_s_wait_asynccnt(6);
 #else
    asm volatile("s_wait_asynccnt 0x6" ::: "memory");
 #endif
#endif
}

__device__ __forceinline__ void wait_async_le0()
{
#if HAVE_ASYNC_LDS
 #if __has_builtin(__builtin_amdgcn_s_wait_asynccnt)
    __builtin_amdgcn_s_wait_asynccnt(0);
 #else
    asm volatile("s_wait_asynccnt 0x0" ::: "memory");
 #endif
#endif
}

// ---------------------------------------------------------------------------
// Dual-GEMM + bias + ReLU on pre-converted bf16 operands:
//   out[M,N] = relu( X1 @ W1^T + X2 @ W2^T + b1 + b2 )
// Block tile 64(M) x 128(N) x 64(K); 8 waves = 2(m) x 4(n); wave tile 32x32.
// Double-buffered LDS: tile t+1 DMA (ASYNCcnt) overlaps tile t WMMAs.
// ---------------------------------------------------------------------------
#define BLK_M 64
#define BLK_N 128
#define BLK_K 64
#define A_TILE (BLK_M * BLK_K)
#define B_TILE (BLK_N * BLK_K)

__global__ __launch_bounds__(256)
void gemm_bf16_relu(const unsigned short* __restrict__ X1, const unsigned short* __restrict__ X2,
                    const unsigned short* __restrict__ W1, const unsigned short* __restrict__ W2,
                    const float* __restrict__ b1, const float* __restrict__ b2,
                    float* __restrict__ out, int N, int K1, int K2)
{
    __shared__ unsigned short As[2 * A_TILE];   // 16 KB (2 x 64x64 bf16)
    __shared__ unsigned short Bs[2 * B_TILE];   // 32 KB (2 x 128x64 bf16)

    const int tid  = threadIdx.x;
    const int lane = tid & 31;
    const int wave = tid >> 5;
    const int wm   = wave & 1;
    const int wn   = wave >> 1;
    const int half = lane >> 4;
    const int l16  = lane & 15;

    const int block_m = blockIdx.y * BLK_M;
    const int block_n = blockIdx.x * BLK_N;

    const int T1 = K1 / BLK_K;
    const int T  = T1 + K2 / BLK_K;

    // issue the 6 async b128 copies (2 A + 4 B) for logical tile t into buffer `buf`
    auto issue_tile = [&](int t, int buf) {
        const unsigned short* X;
        const unsigned short* W;
        int K, kt;
        if (t < T1) { X = X1; W = W1; K = K1; kt = t; }
        else        { X = X2; W = W2; K = K2; kt = t - T1; }
        const int k0 = kt * BLK_K;
        unsigned short* Ab = &As[buf * A_TILE];
        unsigned short* Bb = &Bs[buf * B_TILE];
        #pragma unroll
        for (int it = 0; it < 2; ++it) {
            int u  = tid + it * 256;
            int r  = u >> 3;            // 0..63
            int c8 = (u & 7) * 8;       // 0..56
            copy16_g2l(&X[(size_t)(block_m + r) * K + k0 + c8], &Ab[r * BLK_K + c8]);
        }
        #pragma unroll
        for (int it = 0; it < 4; ++it) {
            int u  = tid + it * 256;
            int r  = u >> 3;            // 0..127
            int c8 = (u & 7) * 8;
            copy16_g2l(&W[(size_t)(block_n + r) * K + k0 + c8], &Bb[r * BLK_K + c8]);
        }
    };

    v8f acc[2][2];
    #pragma unroll
    for (int mt = 0; mt < 2; ++mt)
        #pragma unroll
        for (int nt = 0; nt < 2; ++nt)
            acc[mt][nt] = v8f{};

    issue_tile(0, 0);   // prologue

    #pragma unroll 1
    for (int t = 0; t < T; ++t) {
        const int cur = t & 1;
        // prefetch next tile into the other buffer; its DMA overlaps our WMMAs
        if (t + 1 < T) {
            issue_tile(t + 1, cur ^ 1);
            wait_async_le6();      // tile t (the 6 oldest) complete; t+1 in flight
        } else {
            wait_async_le0();      // drain
        }
        __syncthreads();           // all waves' tile-t data visible in LDS

        const unsigned short* Ab = &As[cur * A_TILE];
        const unsigned short* Bb = &Bs[cur * B_TILE];

        #pragma unroll
        for (int s = 0; s < 2; ++s) {
            FragAB afr[2], bfr[2];
            // A 16x32: lane m=l16; elems 0..7 -> K=half*8+.., 8..15 -> K=16+half*8+..
            #pragma unroll
            for (int mt = 0; mt < 2; ++mt) {
                int row = wm * 32 + mt * 16 + l16;
                afr[mt].q[0] = *(const uint4*)&Ab[row * BLK_K + s * 32 + half * 8];
                afr[mt].q[1] = *(const uint4*)&Ab[row * BLK_K + s * 32 + half * 8 + 16];
            }
            // B 32x16: lane n=l16; elems 0..7 -> K=half*8+.., 8..15 -> K=16+half*8+..
            #pragma unroll
            for (int nt = 0; nt < 2; ++nt) {
                int row = wn * 32 + nt * 16 + l16;
                bfr[nt].q[0] = *(const uint4*)&Bb[row * BLK_K + s * 32 + half * 8];
                bfr[nt].q[1] = *(const uint4*)&Bb[row * BLK_K + s * 32 + half * 8 + 16];
            }
            #pragma unroll
            for (int mt = 0; mt < 2; ++mt)
                #pragma unroll
                for (int nt = 0; nt < 2; ++nt)
                    acc[mt][nt] = __builtin_amdgcn_wmma_f32_16x16x32_bf16(
                        false, afr[mt].v, false, bfr[nt].v,
                        (short)0, acc[mt][nt], false, false);
        }
        __syncthreads();           // everyone done reading buf `cur` before it is reused
    }

    // ---- epilogue: bias + ReLU; C layout: m = half*8 + r, n = l16
    #pragma unroll
    for (int nt = 0; nt < 2; ++nt) {
        int col   = block_n + wn * 32 + nt * 16 + l16;
        float bia = b1[col] + b2[col];
        #pragma unroll
        for (int mt = 0; mt < 2; ++mt) {
            size_t rbase = (size_t)(block_m + wm * 32 + mt * 16 + half * 8);
            float* op = out + rbase * N + col;
            v8f a = acc[mt][nt];
            #pragma unroll
            for (int r = 0; r < 8; ++r) {
                float v = a[r] + bia;
                op[(size_t)r * N] = v > 0.f ? v : 0.f;
            }
        }
    }
}

// ---------------------------------------------------------------------------
// Fallback GEMM on fp32 inputs (conversion in-kernel) — used if ws too small
// ---------------------------------------------------------------------------
__global__ __launch_bounds__(256)
void fused_gemm_relu_f32(const float* __restrict__ X1, const float* __restrict__ X2,
                         const float* __restrict__ W1, const float* __restrict__ W2,
                         const float* __restrict__ b1, const float* __restrict__ b2,
                         float* __restrict__ out, int N, int K1, int K2)
{
    __shared__ unsigned short As[BLK_M * 32];
    __shared__ unsigned short Bs[BLK_N * 32];

    const int tid  = threadIdx.x;
    const int lane = tid & 31;
    const int wave = tid >> 5;
    const int wm   = wave & 1;
    const int wn   = wave >> 1;
    const int half = lane >> 4;
    const int l16  = lane & 15;

    const int block_m = blockIdx.y * BLK_M;
    const int block_n = blockIdx.x * BLK_N;

    v8f acc[2][2];
    #pragma unroll
    for (int mt = 0; mt < 2; ++mt)
        #pragma unroll
        for (int nt = 0; nt < 2; ++nt)
            acc[mt][nt] = v8f{};

    #pragma unroll 1
    for (int ph = 0; ph < 2; ++ph) {
        const float* X = (ph == 0) ? X1 : X2;
        const float* W = (ph == 0) ? W1 : W2;
        const int    K = (ph == 0) ? K1 : K2;

        #pragma unroll 1
        for (int k0 = 0; k0 < K; k0 += 32) {
            #pragma unroll
            for (int it = 0; it < 2; ++it) {
                int f4 = tid + it * 256, r = f4 >> 3, c4 = (f4 & 7) * 4;
                float4 v = *(const float4*)&X[(size_t)(block_m + r) * K + k0 + c4];
                *(uint2*)&As[r * 32 + c4] = pack4_bf16(v);
            }
            #pragma unroll
            for (int it = 0; it < 4; ++it) {
                int f4 = tid + it * 256, r = f4 >> 3, c4 = (f4 & 7) * 4;
                float4 v = *(const float4*)&W[(size_t)(block_n + r) * K + k0 + c4];
                *(uint2*)&Bs[r * 32 + c4] = pack4_bf16(v);
            }
            __syncthreads();

            FragAB afr[2], bfr[2];
            #pragma unroll
            for (int mt = 0; mt < 2; ++mt) {
                int row = wm * 32 + mt * 16 + l16;
                afr[mt].q[0] = *(const uint4*)&As[row * 32 + half * 8];
                afr[mt].q[1] = *(const uint4*)&As[row * 32 + half * 8 + 16];
            }
            #pragma unroll
            for (int nt = 0; nt < 2; ++nt) {
                int row = wn * 32 + nt * 16 + l16;
                bfr[nt].q[0] = *(const uint4*)&Bs[row * 32 + half * 8];
                bfr[nt].q[1] = *(const uint4*)&Bs[row * 32 + half * 8 + 16];
            }
            __syncthreads();

            #pragma unroll
            for (int mt = 0; mt < 2; ++mt)
                #pragma unroll
                for (int nt = 0; nt < 2; ++nt)
                    acc[mt][nt] = __builtin_amdgcn_wmma_f32_16x16x32_bf16(
                        false, afr[mt].v, false, bfr[nt].v,
                        (short)0, acc[mt][nt], false, false);
        }
    }

    #pragma unroll
    for (int nt = 0; nt < 2; ++nt) {
        int col   = block_n + wn * 32 + nt * 16 + l16;
        float bia = b1[col] + b2[col];
        #pragma unroll
        for (int mt = 0; mt < 2; ++mt) {
            size_t rbase = (size_t)(block_m + wm * 32 + mt * 16 + half * 8);
            float* op = out + rbase * N + col;
            v8f a = acc[mt][nt];
            #pragma unroll
            for (int r = 0; r < 8; ++r) {
                float v = a[r] + bia;
                op[(size_t)r * N] = v > 0.f ? v : 0.f;
            }
        }
    }
}

// ---------------------------------------------------------------------------
// Per-node LSTM step (elementwise, C=4). One thread per (b, n).
// ---------------------------------------------------------------------------
__device__ __forceinline__ float sigm(float z) {
    return 1.f / (1.f + __expf(-z));
}

__global__ __launch_bounds__(256)
void lstm_step(const float* __restrict__ x,      // [B, N] pre-activations
               const float* __restrict__ hprev,  // [B, N]
               const float* __restrict__ cprev,  // [N, B, C]
               const float* __restrict__ Wih,    // [N, 4C]
               const float* __restrict__ Whh,    // [N, 4C]
               const float* __restrict__ bias,   // [N, 4C]
               const float* __restrict__ Whr,    // [N, C]
               float* __restrict__ hnew,         // [B, N]
               float* __restrict__ cnew,         // [N, B, C]
               int N)
{
    int idx = blockIdx.x * blockDim.x + threadIdx.x;
    if (idx >= kB * N) return;
    int n = idx % N;
    int b = idx / N;

    float xv = x[(size_t)b * N + n];
    float hv = hprev[(size_t)b * N + n];
    float4 cp = *(const float4*)&cprev[((size_t)n * kB + b) * kC];
    const float cpv[4] = {cp.x, cp.y, cp.z, cp.w};

    const float* wih = &Wih[(size_t)n * 16];
    const float* whh = &Whh[(size_t)n * 16];
    const float* bb  = &bias[(size_t)n * 16];
    const float* whr = &Whr[(size_t)n * kC];

    float hsum = 0.f;
    float cn[4];
    #pragma unroll
    for (int c = 0; c < kC; ++c) {
        float gi = sigm(  xv * wih[c]      + hv * whh[c]      + bb[c]);
        float gf = sigm(  xv * wih[4 + c]  + hv * whh[4 + c]  + bb[4 + c]);
        float gg = tanhf( xv * wih[8 + c]  + hv * whh[8 + c]  + bb[8 + c]);
        float go = sigm(  xv * wih[12 + c] + hv * whh[12 + c] + bb[12 + c]);
        float cc = gf * cpv[c] + gi * gg;
        cn[c] = cc;
        hsum += go * tanhf(cc) * whr[c];
    }
    hnew[(size_t)b * N + n] = hsum;
    float4 co = {cn[0], cn[1], cn[2], cn[3]};
    *(float4*)&cnew[((size_t)n * kB + b) * kC] = co;
}

// ---------------------------------------------------------------------------
// Host-side launch
// ---------------------------------------------------------------------------
extern "C" void kernel_launch(void* const* d_in, const int* in_sizes, int n_in,
                              void* d_out, int out_size, void* d_ws, size_t ws_size,
                              hipStream_t stream)
{
    const float* i_      = (const float*)d_in[0];
    const float* h_      = (const float*)d_in[1];
    const float* o_      = (const float*)d_in[2];
    const float* h_cells = (const float*)d_in[3];
    const float* o_cells = (const float*)d_in[4];
    const float* W_i2h   = (const float*)d_in[5];
    const float* b_i2h   = (const float*)d_in[6];
    const float* W_h2h   = (const float*)d_in[7];
    const float* b_h2h   = (const float*)d_in[8];
    const float* W_i2o   = (const float*)d_in[9];
    const float* b_i2o   = (const float*)d_in[10];
    const float* W_h2o   = (const float*)d_in[11];
    const float* b_h2o   = (const float*)d_in[12];
    const float* Wih_h   = (const float*)d_in[13];
    const float* Whh_h   = (const float*)d_in[14];
    const float* bias_h  = (const float*)d_in[15];
    const float* Whr_h   = (const float*)d_in[16];
    const float* Wih_o   = (const float*)d_in[17];
    const float* Whh_o   = (const float*)d_in[18];
    const float* bias_o  = (const float*)d_in[19];
    const float* Whr_o   = (const float*)d_in[20];

    float* out    = (float*)d_out;
    float* new_h  = out;                                  // B*NH
    float* new_o  = new_h + (size_t)kB * kNH;             // B*NO
    float* new_hc = new_o + (size_t)kB * kNO;             // NH*B*C
    float* new_oc = new_hc + (size_t)kNH * kB * kC;       // NO*B*C

    // workspace layout: [inh f32][ino f32][bf16 copies...]
    const size_t n_i    = (size_t)kB * kNI;
    const size_t n_h    = (size_t)kB * kNH;
    const size_t n_wi2h = (size_t)kNH * kNI;
    const size_t n_wh2h = (size_t)kNH * kNH;
    const size_t n_wi2o = (size_t)kNO * kNI;
    const size_t n_wh2o = (size_t)kNO * kNH;
    const size_t preact_elems = (size_t)kB * kNH + (size_t)kB * kNO;
    const size_t bf_elems = n_i + n_h + n_wi2h + n_wh2h + n_wi2o + n_wh2o;
    const size_t need = preact_elems * 4 + bf_elems * 2;

    float* inh = (float*)d_ws;
    float* ino = inh + (size_t)kB * kNH;

    dim3 blk(256);

    if (ws_size >= need) {
        unsigned short* bfb    = (unsigned short*)(ino + (size_t)kB * kNO);
        unsigned short* i_bf   = bfb;
        unsigned short* h_bf   = i_bf + n_i;
        unsigned short* wi2h_b = h_bf + n_h;
        unsigned short* wh2h_b = wi2h_b + n_wi2h;
        unsigned short* wi2o_b = wh2h_b + n_wh2h;
        unsigned short* wh2o_b = wi2o_b + n_wi2o;

        // one-pass fp32 -> bf16 (weights converted once, not per m-block)
        convert_bf16<<<(int)(n_i    / 8 / 256), blk, 0, stream>>>(i_,    i_bf,   (int)(n_i    / 8));
        convert_bf16<<<(int)(n_h    / 8 / 256), blk, 0, stream>>>(h_,    h_bf,   (int)(n_h    / 8));
        convert_bf16<<<(int)(n_wi2h / 8 / 256), blk, 0, stream>>>(W_i2h, wi2h_b, (int)(n_wi2h / 8));
        convert_bf16<<<(int)(n_wh2h / 8 / 256), blk, 0, stream>>>(W_h2h, wh2h_b, (int)(n_wh2h / 8));
        convert_bf16<<<(int)(n_wi2o / 8 / 256), blk, 0, stream>>>(W_i2o, wi2o_b, (int)(n_wi2o / 8));
        convert_bf16<<<(int)(n_wh2o / 8 / 256), blk, 0, stream>>>(W_h2o, wh2o_b, (int)(n_wh2o / 8));

        gemm_bf16_relu<<<dim3(kNH / BLK_N, kB / BLK_M), blk, 0, stream>>>(
            i_bf, h_bf, wi2h_b, wh2h_b, b_i2h, b_h2h, inh, kNH, kNI, kNH);
        gemm_bf16_relu<<<dim3(kNO / BLK_N, kB / BLK_M), blk, 0, stream>>>(
            i_bf, h_bf, wi2o_b, wh2o_b, b_i2o, b_h2o, ino, kNO, kNI, kNH);
    } else {
        fused_gemm_relu_f32<<<dim3(kNH / BLK_N, kB / BLK_M), blk, 0, stream>>>(
            i_, h_, W_i2h, W_h2h, b_i2h, b_h2h, inh, kNH, kNI, kNH);
        fused_gemm_relu_f32<<<dim3(kNO / BLK_N, kB / BLK_M), blk, 0, stream>>>(
            i_, h_, W_i2o, W_h2o, b_i2o, b_h2o, ino, kNO, kNI, kNH);
    }

    lstm_step<<<(kB * kNH + 255) / 256, blk, 0, stream>>>(
        inh, h_, h_cells, Wih_h, Whh_h, bias_h, Whr_h, new_h, new_hc, kNH);
    lstm_step<<<(kB * kNO + 255) / 256, blk, 0, stream>>>(
        ino, o_, o_cells, Wih_o, Whh_o, bias_o, Whr_o, new_o, new_oc, kNO);
}
